// GatedSpectralConv1d_51376398795171
// MI455X (gfx1250) — matrix operand order repaired and checked by reading, not compiled, not run on hardware
//
#include <hip/hip_runtime.h>
#include <hip/hip_bf16.h>

typedef _Float16 v16h __attribute__((ext_vector_type(16)));
typedef float    v8f  __attribute__((ext_vector_type(8)));
typedef unsigned int u32x4 __attribute__((ext_vector_type(4)));
typedef int          i32x8 __attribute__((ext_vector_type(8)));
typedef int          i32x4 __attribute__((ext_vector_type(4)));

#define WMMA_F16(A, B, C) \
  __builtin_amdgcn_wmma_f32_16x16x32_f16(false, (A), false, (B), (short)0, (C), false, false)

#define PI_F 3.14159265358979323846f

// Problem constants
#define NSAMP 8192
#define MODES 32
#define ROWS  2048      // B*C = 32*64
#define NB    32        // batch
#define NC    64        // channels
#define LSEG  20
#define MCHEB 8
#define LM    160       // LSEG*MCHEB
#define KSPLIT 8

// ---------------------------------------------------------------------------
// Pack a 16x32 f16 A-tile operand (per-lane v16h) from a row-major f32 row.
// A k order used here (contiguous): lanes 0-15 (hi=0): VGPR0-7 = K kc..kc+15
//                  lanes 16-31 (hi=1): VGPR0-7 = K kc+16..kc+31
// ---------------------------------------------------------------------------
__device__ inline v16h pack_a(const float* p, int kc, int hi) {
  const float* q0 = p + kc + 16 * hi;
  const float* q1 = p + kc + 16 * hi + 8;
  v16h r;
#pragma unroll
  for (int j = 0; j < 8; ++j) {
    r[j]     = (_Float16)q0[j];
    r[8 + j] = (_Float16)q1[j];
  }
  return r;
}

// ---------------------------------------------------------------------------
// Setup: CFT combined weight tables Ar/Ai (160 x 32) and interpolation tables.
// Faithful to reference: Ar[l,m,f] = Wr[m,f]*cos(phi[l,f]) + Wi[m,f]*sin(phi[l,f])
// ---------------------------------------------------------------------------
__global__ void k_tables(float* __restrict__ Ar, float* __restrict__ Ai,
                         int* __restrict__ leftIdx, float* __restrict__ wlArr) {
  int t = threadIdx.x;
  float cheb[MCHEB];
  float T[MCHEB][MCHEB];
#pragma unroll
  for (int m = 0; m < MCHEB; ++m)
    cheb[m] = -cosf((2.0f * m + 1.0f) * PI_F / (2.0f * MCHEB));
#pragma unroll
  for (int k = 0; k < MCHEB; ++k)
#pragma unroll
    for (int m = 0; m < MCHEB; ++m)
      T[k][m] = cosf((float)k * acosf(cheb[m]));

  for (int idx = t; idx < LM * MODES; idx += blockDim.x) {
    int lm = idx >> 5;          // 0..159
    int f  = idx & 31;
    int l  = lm / MCHEB;
    int m  = lm % MCHEB;
    float wr = 0.f, wi = 0.f;
    for (int mm = 0; mm < MCHEB; ++mm) {
      float ang = cheb[mm] * (float)f * (PI_F / (float)LSEG);
      wr += T[m][mm] * cosf(ang);
      wi += T[m][mm] * (-sinf(ang));
    }
    wr *= 1.0f / (2.0f * LSEG);
    wi *= 1.0f / (2.0f * LSEG);
    float phi = 2.0f * PI_F * ((float)l / (float)LSEG) * (float)f;
    float cs = cosf(phi), sn = sinf(phi);
    Ar[idx] = wr * cs + wi * sn;
    Ai[idx] = -wr * sn + wi * cs;
  }

  if (t < LM) {
    int l = t / MCHEB, m = t % MCHEB;
    float ch = -cosf((2.0f * m + 1.0f) * PI_F / (2.0f * MCHEB));
    float ts = (float)l / (float)LSEG + (1.0f / (2.0f * LSEG)) * (ch + 1.0f);
    float pos = ts * (float)(NSAMP - 1);
    int right = (int)ceilf(pos - 1e-6f);
    if (right < 1) right = 1;
    if (right > NSAMP - 1) right = NSAMP - 1;
    int left = right - 1;
    float tl = (float)left / (float)(NSAMP - 1);
    float tr = (float)right / (float)(NSAMP - 1);
    float denom = (tr - tl == 0.f) ? 1.f : (tr - tl);
    float w_r = (ts - tl) / denom;
    leftIdx[t] = left;
    wlArr[t]   = 1.0f - w_r;
  }
}

// ---------------------------------------------------------------------------
// Setup: DFT twiddle matrix in WMMA B-register layout, f16.
// ---------------------------------------------------------------------------
__global__ void k_dftw(_Float16* __restrict__ w) {
  int idx = blockIdx.x * blockDim.x + threadIdx.x;
  if (idx >= 4 * 256 * 32 * 16) return;
  int j    = idx & 15;
  int lane = (idx >> 4) & 31;
  int kt   = (idx >> 9) & 255;
  int nt   = idx >> 17;
  int hi   = lane >> 4;
  int c    = nt * 16 + (lane & 15);
  int k    = kt * 32 + 16 * hi + j;       // sample index n
  int f    = (c < 32) ? c : (c - 32);
  int ph   = (f * k) & (NSAMP - 1);       // (f*n) mod 8192
  float ang = 2.0f * PI_F * (float)ph / (float)NSAMP;
  float v = (c < 32) ? cosf(ang) : -sinf(ang);
  w[idx] = (_Float16)v;
}

// ---------------------------------------------------------------------------
// Setup: iDFT matrix in WMMA B-register layout, f16 (1/N, 2x for f>0 folded).
// ---------------------------------------------------------------------------
__global__ void k_idftw(_Float16* __restrict__ w) {
  int idx = blockIdx.x * blockDim.x + threadIdx.x;
  if (idx >= 512 * 2 * 32 * 16) return;
  int j    = idx & 15;
  int lane = (idx >> 4) & 31;
  int kt   = (idx >> 9) & 1;
  int nt   = idx >> 10;
  int hi   = lane >> 4;
  int n    = nt * 16 + (lane & 15);       // output sample
  int k    = kt * 32 + 16 * hi + j;       // K index (0..63)
  int f    = (k < 32) ? k : (k - 32);
  float s  = (f == 0) ? 1.0f : 2.0f;
  int ph   = (f * n) & (NSAMP - 1);
  float ang = 2.0f * PI_F * (float)ph / (float)NSAMP;
  float v = (k < 32) ? (s * cosf(ang) / (float)NSAMP)
                     : (-s * sinf(ang) / (float)NSAMP);
  w[idx] = (_Float16)v;
}

// ---------------------------------------------------------------------------
// Truncated DFT GEMM: (2048 x 8192) f32 -> (2048 x 64) via f16 WMMA.
// grid = (128 m-tiles, 8 k-splits), block = 32. Deterministic k-split partials.
// ---------------------------------------------------------------------------
__global__ void __launch_bounds__(32)
k_dft(const float* __restrict__ x, const _Float16* __restrict__ dftW,
      float* __restrict__ part) {
  int mt   = blockIdx.x;            // 0..127
  int ks   = blockIdx.y;            // 0..7
  int lane = threadIdx.x;
  int hi   = lane >> 4;
  int nloc = lane & 15;
  const float* rowPtr = x + (size_t)(mt * 16 + nloc) * NSAMP;

  v8f acc[4] = {};
  int kt0 = ks * 32;
  for (int kt = kt0; kt < kt0 + 32; ++kt) {
    int kc = kt * 32;
    __builtin_prefetch(rowPtr + kc + 128, 0, 0);   // global_prefetch_b8
    v16h a = pack_a(rowPtr, kc, hi);
#pragma unroll
    for (int nt = 0; nt < 4; ++nt) {
      const v16h b = *(const v16h*)(dftW + ((size_t)(nt * 256 + kt) * 32 + lane) * 16);
      acc[nt] = WMMA_F16(a, b, acc[nt]);
    }
  }
  float* slab = part + (size_t)ks * ROWS * 64;
#pragma unroll
  for (int nt = 0; nt < 4; ++nt)
#pragma unroll
    for (int v = 0; v < 8; ++v)
      slab[(size_t)(mt * 16 + v + 8 * hi) * 64 + nt * 16 + nloc] = acc[nt][v];
}

__global__ void k_dft_reduce(const float* __restrict__ part, float* __restrict__ cdft) {
  int i = blockIdx.x * blockDim.x + threadIdx.x;
  if (i >= ROWS * 64) return;
  float s = 0.f;
#pragma unroll
  for (int k = 0; k < KSPLIT; ++k) s += part[(size_t)k * ROWS * 64 + i];
  cdft[i] = s;
}

// ---------------------------------------------------------------------------
// CFT magnitude + LayerNorm. One wave per (b,c) row; lane == mode.
// ---------------------------------------------------------------------------
__global__ void __launch_bounds__(32)
k_cft(const float* __restrict__ x, const float* __restrict__ Ar,
      const float* __restrict__ Ai, const int* __restrict__ leftIdx,
      const float* __restrict__ wlArr, const float* __restrict__ gamma,
      const float* __restrict__ beta, float* __restrict__ gnorm) {
  __shared__ float seg[LM];
  int row  = blockIdx.x;
  int lane = threadIdx.x;
  const float* rp = x + (size_t)row * NSAMP;
  for (int i = lane; i < LM; i += 32) {
    int li = leftIdx[i];
    float wl = wlArr[i];
    seg[i] = wl * rp[li] + (1.0f - wl) * rp[li + 1];
  }
  __syncthreads();
  int f = lane;
  float re = 0.f, im = 0.f;
  for (int lm = 0; lm < LM; ++lm) {
    float s = seg[lm];
    re += s * Ar[lm * MODES + f];
    im += s * Ai[lm * MODES + f];
  }
  float mag = sqrtf(re * re + im * im);
  float mu = mag;
#pragma unroll
  for (int o = 16; o; o >>= 1) mu += __shfl_xor(mu, o, 32);
  mu *= (1.0f / 32.0f);
  float d = mag - mu;
  float var = d * d;
#pragma unroll
  for (int o = 16; o; o >>= 1) var += __shfl_xor(var, o, 32);
  var *= (1.0f / 32.0f);
  gnorm[(size_t)row * MODES + f] = d * rsqrtf(var + 1e-5f) * gamma[f] + beta[f];
}

// ---------------------------------------------------------------------------
// Gate: sigmoid(gate_w @ gnorm) per batch. M=64(o), K=64(i), N=32(f). f16 WMMA.
// ---------------------------------------------------------------------------
__global__ void __launch_bounds__(32)
k_gate(const float* __restrict__ gw, const float* __restrict__ gnorm,
       float* __restrict__ gate) {
  int b    = blockIdx.y;
  int mt   = blockIdx.x >> 1;
  int ntf  = blockIdx.x & 1;
  int lane = threadIdx.x;
  int hi   = lane >> 4;
  int nloc = lane & 15;
  v8f acc = {};
#pragma unroll
  for (int kc = 0; kc < 64; kc += 32) {
    v16h a = pack_a(gw + (size_t)(mt * 16 + nloc) * NC, kc, hi);
    v16h bb;
#pragma unroll
    for (int j = 0; j < 16; ++j) {
      int k = kc + 16 * hi + j;
      bb[j] = (_Float16)gnorm[((size_t)b * NC + k) * MODES + ntf * 16 + nloc];
    }
    acc = WMMA_F16(a, bb, acc);
  }
#pragma unroll
  for (int v = 0; v < 8; ++v) {
    float g = acc[v];
    gate[((size_t)b * NC + mt * 16 + v + 8 * hi) * MODES + ntf * 16 + nloc] =
        1.0f / (1.0f + expf(-g));
  }
}

// gated = x_fft[:, :, :32] * gate (complex * real), split re/im
__global__ void k_gated(const float* __restrict__ cdft, const float* __restrict__ gate,
                        float* __restrict__ gre, float* __restrict__ gim) {
  int i = blockIdx.x * blockDim.x + threadIdx.x;
  if (i >= ROWS * MODES) return;
  int row = i >> 5, f = i & 31;
  float g = gate[i];
  gre[i] = cdft[(size_t)row * 64 + f] * g;
  gim[i] = cdft[(size_t)row * 64 + 32 + f] * g;
}

// ---------------------------------------------------------------------------
// Complex per-mode channel mix: out[b,o] = sum_i gated[b,i] * W1[i,o] (per f).
// ---------------------------------------------------------------------------
__global__ void __launch_bounds__(32)
k_mix(const float* __restrict__ gre, const float* __restrict__ gim,
      const float* __restrict__ w1r, const float* __restrict__ w1i,
      float* __restrict__ cmix) {
  int f    = blockIdx.y;
  int bt   = blockIdx.x >> 2;
  int nt   = blockIdx.x & 3;
  int lane = threadIdx.x;
  int hi   = lane >> 4;
  int nloc = lane & 15;
  int bm   = bt * 16 + nloc;           // batch index for this lane's A rows
  v8f accR = {}, accI = {};
#pragma unroll
  for (int kc = 0; kc < 64; kc += 32) {
    v16h ar, aip, ain;
#pragma unroll
    for (int j = 0; j < 16; ++j) {
      int k = kc + 16 * hi + j;
      float xr = gre[((size_t)bm * NC + k) * MODES + f];
      float xi = gim[((size_t)bm * NC + k) * MODES + f];
      ar[j]  = (_Float16)xr;
      aip[j] = (_Float16)xi;
      ain[j] = (_Float16)(-xi);
    }
    v16h br, bi;
#pragma unroll
    for (int j = 0; j < 16; ++j) {
      int k = kc + 16 * hi + j;
      br[j] = (_Float16)w1r[((size_t)k * NC + nt * 16 + nloc) * MODES + f];
      bi[j] = (_Float16)w1i[((size_t)k * NC + nt * 16 + nloc) * MODES + f];
    }
    accR = WMMA_F16(ar, br, accR);
    accR = WMMA_F16(ain, bi, accR);   // - gi*Wi
    accI = WMMA_F16(ar, bi, accI);
    accI = WMMA_F16(aip, br, accI);   // + gi*Wr
  }
#pragma unroll
  for (int v = 0; v < 8; ++v) {
    int b = bt * 16 + v + 8 * hi;
    int o = nt * 16 + nloc;
    cmix[((size_t)b * NC + o) * 64 + f]      = accR[v];
    cmix[((size_t)b * NC + o) * 64 + 32 + f] = accI[v];
  }
}

// ---------------------------------------------------------------------------
// Truncated iDFT GEMM with TDM staging: (2048 x 64) x (64 x 8192) -> out.
// grid = (128 m-tiles, 32 n-groups of 16 n-tiles), block = 256 (8 waves).
// Wave 0 DMAs a 32 KB contiguous slab of pre-layouted B-tiles into LDS via
// TENSOR_LOAD_TO_LDS (1-D tile, 8-byte elements), waits on TENSORcnt, then
// all 8 waves consume it from LDS (2 n-tiles each, 2 WMMAs per n-tile).
// ---------------------------------------------------------------------------
__global__ void __launch_bounds__(256)
k_idft(const float* __restrict__ cmix, const _Float16* __restrict__ idftW,
       float* __restrict__ out) {
  __shared__ _Float16 bbuf[16 * 1024];   // 16 n-tiles x 2 KB = 32 KB
  int mt   = blockIdx.x;                 // 0..127
  int ng   = blockIdx.y;                 // 0..31
  int tid  = threadIdx.x;
  int wave = tid >> 5;
  int lane = tid & 31;
  int hi   = lane >> 4;
  int nloc = lane & 15;

  if (wave == 0) {
    // D# descriptor (ISA 8.3-8.6): 1-D tile of 4096 x 8-byte elements.
    unsigned long long ga =
        (unsigned long long)(uintptr_t)idftW + (unsigned long long)ng * 32768ull;
    unsigned int ldsAddr = (unsigned int)(uintptr_t)(void*)bbuf;  // low 32b = LDS offset
    u32x4 g0;
    g0[0] = 1u;                                   // count=1, user descriptor
    g0[1] = ldsAddr;                              // lds_addr
    g0[2] = (unsigned int)(ga & 0xffffffffu);     // global_addr[31:0]
    g0[3] = (unsigned int)((ga >> 32) & 0x01ffffffu) | (2u << 30);  // addr[56:32], type=2
    i32x8 g1;
    g1[0] = (int)(3u << 16);          // workgroup_mask=0, data_size=3 (8B)
    g1[1] = (int)(4096u << 16);       // tensor_dim0 low16 = 4096
    g1[2] = (int)(1u << 16);          // tensor_dim0 hi=0, tensor_dim1 low16 = 1
    g1[3] = (int)(4096u << 16);       // tensor_dim1 hi=0, tile_dim0 = 4096
    g1[4] = 0;                        // tile_dim1=0 (unused), tile_dim2=0
    g1[5] = 4096;                     // tensor_dim0_stride low 32
    g1[6] = 0;
    g1[7] = 0;
    i32x4 g2 = {0, 0, 0, 0};
    i32x4 g3 = {0, 0, 0, 0};
    i32x8 g4 = {0, 0, 0, 0, 0, 0, 0, 0};   // extra descriptor group (clang-23 6-arg form)
    __builtin_amdgcn_tensor_load_to_lds(g0, g1, g2, g3, g4, 0);
    __builtin_amdgcn_s_wait_tensorcnt(0);
  }
  __syncthreads();

  const float* arow = cmix + (size_t)(mt * 16 + nloc) * 64;
  v16h a0 = pack_a(arow, 0, hi);
  v16h a1 = pack_a(arow, 32, hi);

#pragma unroll
  for (int e = 0; e < 2; ++e) {
    int ntl = wave * 2 + e;            // local n-tile 0..15
    int nt  = ng * 16 + ntl;           // global n-tile 0..511
    const v16h b0 = *(const v16h*)(bbuf + ((size_t)(ntl * 2 + 0) * 32 + lane) * 16);
    const v16h b1 = *(const v16h*)(bbuf + ((size_t)(ntl * 2 + 1) * 32 + lane) * 16);
    v8f acc = {};
    acc = WMMA_F16(a0, b0, acc);
    acc = WMMA_F16(a1, b1, acc);
#pragma unroll
    for (int v = 0; v < 8; ++v)
      out[(size_t)(mt * 16 + v + 8 * hi) * NSAMP + nt * 16 + nloc] = acc[v];
  }
}

// ---------------------------------------------------------------------------
extern "C" void kernel_launch(void* const* d_in, const int* in_sizes, int n_in,
                              void* d_out, int out_size, void* d_ws, size_t ws_size,
                              hipStream_t stream) {
  const float* x     = (const float*)d_in[0];   // (32, 64, 8192)
  const float* w1r   = (const float*)d_in[1];   // (64, 64, 32)
  const float* w1i   = (const float*)d_in[2];   // (64, 64, 32)
  const float* gw    = (const float*)d_in[3];   // (64, 64)
  const float* gamma = (const float*)d_in[4];   // (32,)
  const float* beta  = (const float*)d_in[5];   // (32,)
  float* out = (float*)d_out;                   // (32, 64, 8192)

  char* ws = (char*)d_ws;
  size_t off = 0;
  _Float16* dftW  = (_Float16*)(ws + off); off += (size_t)4 * 256 * 32 * 16 * 2;   // 1 MB
  _Float16* idftW = (_Float16*)(ws + off); off += (size_t)512 * 2 * 32 * 16 * 2;   // 1 MB
  float* part  = (float*)(ws + off); off += (size_t)KSPLIT * ROWS * 64 * 4;        // 4 MB
  float* cdft  = (float*)(ws + off); off += (size_t)ROWS * 64 * 4;
  float* gnorm = (float*)(ws + off); off += (size_t)ROWS * MODES * 4;
  float* gate  = (float*)(ws + off); off += (size_t)ROWS * MODES * 4;
  float* gre   = (float*)(ws + off); off += (size_t)ROWS * MODES * 4;
  float* gim   = (float*)(ws + off); off += (size_t)ROWS * MODES * 4;
  float* cmix  = (float*)(ws + off); off += (size_t)ROWS * 64 * 4;
  float* Ar    = (float*)(ws + off); off += (size_t)LM * MODES * 4;
  float* Ai    = (float*)(ws + off); off += (size_t)LM * MODES * 4;
  int*   leftI = (int*)(ws + off);   off += (size_t)LM * 4;
  float* wlA   = (float*)(ws + off); off += (size_t)LM * 4;

  // 1. Constant tables (recomputed every launch: deterministic)
  k_tables<<<1, 256, 0, stream>>>(Ar, Ai, leftI, wlA);
  k_dftw<<<(4 * 256 * 32 * 16 + 255) / 256, 256, 0, stream>>>(dftW);
  k_idftw<<<(512 * 2 * 32 * 16 + 255) / 256, 256, 0, stream>>>(idftW);

  // 2. Truncated DFT (WMMA f16, K-split partials, deterministic reduce)
  k_dft<<<dim3(128, KSPLIT), 32, 0, stream>>>(x, dftW, part);
  k_dft_reduce<<<(ROWS * 64 + 255) / 256, 256, 0, stream>>>(part, cdft);

  // 3. CFT magnitude + LayerNorm
  k_cft<<<ROWS, 32, 0, stream>>>(x, Ar, Ai, leftI, wlA, gamma, beta, gnorm);

  // 4. Gate GEMM + sigmoid (WMMA f16)
  k_gate<<<dim3(8, NB), 32, 0, stream>>>(gw, gnorm, gate);

  // 5. Apply gate to spectrum
  k_gated<<<(ROWS * MODES + 255) / 256, 256, 0, stream>>>(cdft, gate, gre, gim);

  // 6. Complex per-mode channel mixing (WMMA f16)
  k_mix<<<dim3(8, MODES), 32, 0, stream>>>(gre, gim, w1r, w1i, cmix);

  // 7. Truncated iDFT -> output (WMMA f16 + TDM-staged B tiles in LDS)
  k_idft<<<dim3(128, 32), 256, 0, stream>>>(cmix, idftW, out);
}
